// OnlyAttention_65919158059581
// MI455X (gfx1250) — compile-verified, matrix operand order checked by reading, not run on hardware
//
#include <hip/hip_runtime.h>

#define D_IN  1024
#define SEQ   2048
#define BATCH 4
#define NH    16
#define DH    64
#define BS    (BATCH * SEQ)   // 8192

typedef __attribute__((ext_vector_type(16))) __bf16 v16bf;
typedef __attribute__((ext_vector_type(8)))  float  v8f;

union Frag { v16bf v; uint4 q[2]; };

__device__ __forceinline__ unsigned short f2bf(float f) {
  unsigned int u = __float_as_uint(f);
  u += 0x7FFFu + ((u >> 16) & 1u);   // round-to-nearest-even
  return (unsigned short)(u >> 16);
}

// Load one 16x32 bf16 A/B fragment (8 VGPRs) for this lane.
// Packing per ISA 7.12.2: lanes 0-15 hold K = {k0..k0+7, k0+16..k0+23},
// lanes 16-31 hold K shifted by +8.  e0 already includes the +8 half shift.
__device__ __forceinline__ v16bf load_frag_g(const unsigned short* p, long long e0) {
  Frag f;
  f.q[0] = *reinterpret_cast<const uint4*>(p + e0);
  f.q[1] = *reinterpret_cast<const uint4*>(p + e0 + 16);
  return f.v;
}

// ---------------------------------------------------------------------------
// Kernel 1: fp32 -> bf16 for x; fp32 -> bf16 *transposed* for Wq/Wk/Wv.
// ---------------------------------------------------------------------------
__global__ void cvt_kernel(const float* __restrict__ x,
                           const float* __restrict__ wq,
                           const float* __restrict__ wk,
                           const float* __restrict__ wv,
                           unsigned short* __restrict__ xb,
                           unsigned short* __restrict__ wt) {
  long long idx = (long long)blockIdx.x * blockDim.x + threadIdx.x;
  const long long NX = (long long)BS * D_IN;
  if (idx < NX) { xb[idx] = f2bf(x[idx]); return; }
  long long j = idx - NX;
  if (j >= 3LL * D_IN * D_IN) return;
  int w = (int)(j / (D_IN * D_IN));
  int r = (int)(j % (D_IN * D_IN));
  int k = r / D_IN, n = r % D_IN;
  const float* W = (w == 0) ? wq : (w == 1) ? wk : wv;
  wt[(long long)w * D_IN * D_IN + (long long)n * D_IN + k] =
      f2bf(W[(long long)k * D_IN + n]);
}

// ---------------------------------------------------------------------------
// Kernel 2: projections.  One wave computes a 16(M) x 64(N) output strip:
// 4 accumulators sharing one A fragment per K-step (4x A reuse, 128 WMMA/wave).
// Since the N-supertile is 64 wide == head dim, h is constant per wave.
// proj 0/1 -> Q/K [B,H,S,64] (Q pre-scaled by 1/sqrt(Dh)); proj 2 -> V^T [B,H,64,S].
// ---------------------------------------------------------------------------
__global__ void __launch_bounds__(256) proj_kernel(
    const unsigned short* __restrict__ xb,
    const unsigned short* __restrict__ wt,
    const float* __restrict__ bq,
    const float* __restrict__ bk,
    const float* __restrict__ bv,
    unsigned short* __restrict__ Qp,
    unsigned short* __restrict__ Kp,
    unsigned short* __restrict__ Vt) {
  const int lane  = threadIdx.x & 31;
  const int wslot = threadIdx.x >> 5;
  const int w     = blockIdx.x * 8 + wslot;    // 0 .. 3*512*16-1
  const int proj  = w / (512 * 16);
  const int rem   = w % (512 * 16);
  const int mt    = rem >> 4;                  // 0..511  (M tiles of 16)
  const int nt4   = rem & 15;                  // 0..15   (N strips of 64 == head)
  const int rlo   = lane & 15;
  const int khalf = (lane >> 4) * 8;

  const unsigned short* Wb   = wt + (long long)proj * D_IN * D_IN;
  const float*          bias = (proj == 0) ? bq : (proj == 1) ? bk : bv;

  const int arow  = mt * 16 + rlo;             // A rows: both halves map M=rlo
  const long long brow0 = (long long)(nt4 * 64 +  0 + rlo) * D_IN;
  const long long brow1 = (long long)(nt4 * 64 + 16 + rlo) * D_IN;
  const long long brow2 = (long long)(nt4 * 64 + 32 + rlo) * D_IN;
  const long long brow3 = (long long)(nt4 * 64 + 48 + rlo) * D_IN;

  v8f c0 = {}, c1 = {}, c2 = {}, c3 = {};
  for (int kk = 0; kk < D_IN; kk += 32) {
    const long long ko = kk + khalf;
    v16bf a  = load_frag_g(xb, (long long)arow * D_IN + ko);
    v16bf b0 = load_frag_g(Wb, brow0 + ko);
    v16bf b1 = load_frag_g(Wb, brow1 + ko);
    v16bf b2 = load_frag_g(Wb, brow2 + ko);
    v16bf b3 = load_frag_g(Wb, brow3 + ko);
    c0 = __builtin_amdgcn_wmma_f32_16x16x32_bf16(false, a, false, b0, (short)0, c0, false, false);
    c1 = __builtin_amdgcn_wmma_f32_16x16x32_bf16(false, a, false, b1, (short)0, c1, false, false);
    c2 = __builtin_amdgcn_wmma_f32_16x16x32_bf16(false, a, false, b2, (short)0, c2, false, false);
    c3 = __builtin_amdgcn_wmma_f32_16x16x32_bf16(false, a, false, b3, (short)0, c3, false, false);
  }

  const int h = nt4;                           // constant head for this wave
  const float qscale = (proj == 0) ? 0.125f : 1.0f;  // fold 1/sqrt(64) into Q
  float bias4[4];
  bias4[0] = bias[h * 64 +  0 + rlo];
  bias4[1] = bias[h * 64 + 16 + rlo];
  bias4[2] = bias[h * 64 + 32 + rlo];
  bias4[3] = bias[h * 64 + 48 + rlo];

  for (int r = 0; r < 8; r++) {
    int m  = mt * 16 + r + ((lane >> 4) << 3); // C layout row
    int b_ = m >> 11;                          // / SEQ
    int s  = m & (SEQ - 1);
    long long bh = (long long)b_ * NH + h;
    float v0 = (c0[r] + bias4[0]) * qscale;
    float v1 = (c1[r] + bias4[1]) * qscale;
    float v2 = (c2[r] + bias4[2]) * qscale;
    float v3 = (c3[r] + bias4[3]) * qscale;
    if (proj == 2) {                           // V^T [B,H,64,S]
      long long vb = (bh * DH) * SEQ + s;
      Vt[vb + (long long)( 0 + rlo) * SEQ] = f2bf(v0);
      Vt[vb + (long long)(16 + rlo) * SEQ] = f2bf(v1);
      Vt[vb + (long long)(32 + rlo) * SEQ] = f2bf(v2);
      Vt[vb + (long long)(48 + rlo) * SEQ] = f2bf(v3);
    } else {                                   // Q/K [B,H,S,64]
      unsigned short* P = (proj == 0) ? Qp : Kp;
      long long qb = (bh * SEQ + s) * DH;
      P[qb +  0 + rlo] = f2bf(v0);
      P[qb + 16 + rlo] = f2bf(v1);
      P[qb + 32 + rlo] = f2bf(v2);
      P[qb + 48 + rlo] = f2bf(v3);
    }
  }
}

// ---------------------------------------------------------------------------
// Kernel 3: flash attention.  One wave = 16 query rows x full Dh=64.
// Streams K/V in 32-column strips; online softmax; P relaid to A-format via LDS.
// Q is pre-scaled by 1/sqrt(Dh) at projection time.
// ---------------------------------------------------------------------------
__global__ void __launch_bounds__(256) attn_kernel(
    const unsigned short* __restrict__ Qp,
    const unsigned short* __restrict__ Kp,
    const unsigned short* __restrict__ Vt,
    float* __restrict__ out) {
  __shared__ unsigned short plds[8 * 16 * 32];   // 8 KB: 1 KB P tile per wave

  const int lane  = threadIdx.x & 31;
  const int wslot = threadIdx.x >> 5;
  const int w     = blockIdx.x * 8 + wslot;      // 0..8191
  const int bh    = w >> 7;                      // (b*NH + h)
  const int qt    = w & 127;                     // Q tile of 16 rows
  const int rlo   = lane & 15;
  const int hi    = lane >> 4;
  const int khalf = hi * 8;

  const unsigned short* Q = Qp + (long long)bh * SEQ * DH;
  const unsigned short* K = Kp + (long long)bh * SEQ * DH;
  const unsigned short* V = Vt + (long long)bh * DH * SEQ;
  unsigned short* myp = plds + wslot * (16 * 32);

  // Q fragments (reused all iterations): k 0..31 and 32..63
  const int qrow = qt * 16 + rlo;
  const v16bf aq0 = load_frag_g(Q, (long long)qrow * DH + 0  + khalf);
  const v16bf aq1 = load_frag_g(Q, (long long)qrow * DH + 32 + khalf);

  v8f acc0 = {}, acc1 = {}, acc2 = {}, acc3 = {};
  float mrow[8], lrow[8];
  for (int r = 0; r < 8; r++) { mrow[r] = -1e30f; lrow[r] = 0.0f; }

  for (int t0 = 0; t0 < SEQ; t0 += 32) {
    if (t0 + 32 < SEQ) {
      __builtin_prefetch(K + (long long)(t0 + 32) * DH, 0, 0);
      __builtin_prefetch(V + (long long)rlo * SEQ + t0 + 32, 0, 0);
    }

    // --- K fragments for two 16x16 score tiles covering kv cols t0..t0+31 ---
    v16bf bk00 = load_frag_g(K, (long long)(t0 +      rlo) * DH + 0  + khalf);
    v16bf bk01 = load_frag_g(K, (long long)(t0 +      rlo) * DH + 32 + khalf);
    v16bf bk10 = load_frag_g(K, (long long)(t0 + 16 + rlo) * DH + 0  + khalf);
    v16bf bk11 = load_frag_g(K, (long long)(t0 + 16 + rlo) * DH + 32 + khalf);
    // --- V fragments hoisted: fetched under the softmax VALU work ---
    v16bf bv0 = load_frag_g(V, (long long)( 0 + rlo) * SEQ + t0 + khalf);
    v16bf bv1 = load_frag_g(V, (long long)(16 + rlo) * SEQ + t0 + khalf);
    v16bf bv2 = load_frag_g(V, (long long)(32 + rlo) * SEQ + t0 + khalf);
    v16bf bv3 = load_frag_g(V, (long long)(48 + rlo) * SEQ + t0 + khalf);

    v8f s0 = {}, s1 = {};
    s0 = __builtin_amdgcn_wmma_f32_16x16x32_bf16(false, aq0, false, bk00, (short)0, s0, false, false);
    s0 = __builtin_amdgcn_wmma_f32_16x16x32_bf16(false, aq1, false, bk01, (short)0, s0, false, false);
    s1 = __builtin_amdgcn_wmma_f32_16x16x32_bf16(false, aq0, false, bk10, (short)0, s1, false, false);
    s1 = __builtin_amdgcn_wmma_f32_16x16x32_bf16(false, aq1, false, bk11, (short)0, s1, false, false);

    // --- online softmax over the 16x32 strip; write P (bf16) to LDS ---
    for (int r = 0; r < 8; r++) {
      float a0 = s0[r];
      float a1 = s1[r];
      float mx = fmaxf(a0, a1);
      for (int msk = 8; msk >= 1; msk >>= 1)      // row reduce within 16-lane half
        mx = fmaxf(mx, __shfl_xor(mx, msk, 32));
      float mnew = fmaxf(mrow[r], mx);
      float corr = __expf(mrow[r] - mnew);
      float p0 = __expf(a0 - mnew);
      float p1 = __expf(a1 - mnew);
      float sum = p0 + p1;
      for (int msk = 8; msk >= 1; msk >>= 1)
        sum += __shfl_xor(sum, msk, 32);
      mrow[r] = mnew;
      lrow[r] = lrow[r] * corr + sum;
      acc0[r] *= corr; acc1[r] *= corr; acc2[r] *= corr; acc3[r] *= corr;
      int row = r + (hi << 3);                    // C-layout row
      myp[row * 32 + rlo]      = f2bf(p0);
      myp[row * 32 + rlo + 16] = f2bf(p1);
    }
    __builtin_amdgcn_wave_barrier();
    asm volatile("s_wait_dscnt 0x0" ::: "memory");

    // --- reload P in A-format (lane transpose via LDS) ---
    Frag fp;
    {
      const unsigned short* pb = myp + rlo * 32 + khalf;
      fp.q[0] = *reinterpret_cast<const uint4*>(pb);
      fp.q[1] = *reinterpret_cast<const uint4*>(pb + 16);
    }
    asm volatile("s_wait_dscnt 0x0" ::: "memory");
    __builtin_amdgcn_wave_barrier();
    v16bf ap = fp.v;

    // --- PV: acc[j] += P(16x32) * V(32x16_j), V^T rows are contiguous in t ---
    acc0 = __builtin_amdgcn_wmma_f32_16x16x32_bf16(false, ap, false, bv0, (short)0, acc0, false, false);
    acc1 = __builtin_amdgcn_wmma_f32_16x16x32_bf16(false, ap, false, bv1, (short)0, acc1, false, false);
    acc2 = __builtin_amdgcn_wmma_f32_16x16x32_bf16(false, ap, false, bv2, (short)0, acc2, false, false);
    acc3 = __builtin_amdgcn_wmma_f32_16x16x32_bf16(false, ap, false, bv3, (short)0, acc3, false, false);
  }

  // --- epilogue: normalize and store [B, S, H*Dh] fp32 ---
  const int b_ = bh >> 4, h = bh & 15;
  for (int r = 0; r < 8; r++) {
    int srow = qt * 16 + r + (hi << 3);
    float inv = 1.0f / lrow[r];
    long long base = ((long long)(b_ * SEQ + srow)) * (NH * DH) + h * DH;
    out[base +  0 + rlo] = acc0[r] * inv;
    out[base + 16 + rlo] = acc1[r] * inv;
    out[base + 32 + rlo] = acc2[r] * inv;
    out[base + 48 + rlo] = acc3[r] * inv;
  }
}

// ---------------------------------------------------------------------------
extern "C" void kernel_launch(void* const* d_in, const int* in_sizes, int n_in,
                              void* d_out, int out_size, void* d_ws, size_t ws_size,
                              hipStream_t stream) {
  const float* x  = (const float*)d_in[0];
  const float* Wq = (const float*)d_in[1];
  const float* bq = (const float*)d_in[2];
  const float* Wk = (const float*)d_in[3];
  const float* bk = (const float*)d_in[4];
  const float* Wv = (const float*)d_in[5];
  const float* bv = (const float*)d_in[6];
  float* out = (float*)d_out;
  char* ws = (char*)d_ws;

  // Workspace layout (bytes):
  //   xb : bf16 x                     [8192,1024]      16 MB  @ 0
  //   wt : bf16 W^T (q,k,v)           3x[1024,1024]     6 MB  @ 16 MB
  //   Qp : bf16 Q  [B,H,S,64]                          16 MB  @ 22 MB
  //   Kp : bf16 K  [B,H,S,64]                          16 MB  @ 38 MB
  //   Vt : bf16 V^T [B,H,64,S]                         16 MB  @ 54 MB
  unsigned short* xb = (unsigned short*)(ws);
  unsigned short* wt = (unsigned short*)(ws + 16777216LL);
  unsigned short* Qp = (unsigned short*)(ws + 23068672LL);
  unsigned short* Kp = (unsigned short*)(ws + 39845888LL);
  unsigned short* Vt = (unsigned short*)(ws + 56623104LL);

  const long long ncvt = (long long)BS * D_IN + 3LL * D_IN * D_IN;  // 11,534,336
  const int cvtBlocks = (int)((ncvt + 255) / 256);
  cvt_kernel<<<cvtBlocks, 256, 0, stream>>>(x, Wq, Wk, Wv, xb, wt);

  // 3 projections * 512 M-tiles * 16 N-strips = 24576 waves / 8 per block
  proj_kernel<<<24576 / 8, 256, 0, stream>>>(xb, wt, bq, bk, bv, Qp, Kp, Vt);

  // B*H*(S/16) = 8192 waves / 8 per block
  attn_kernel<<<8192 / 8, 256, 0, stream>>>(Qp, Kp, Vt, out);
}
